// TransformerBlock_19731079758371
// MI455X (gfx1250) — matrix operand order repaired and checked by reading, not compiled, not run on hardware
//
#include <hip/hip_runtime.h>
#include <hip/hip_bf16.h>
#include <math.h>
#include <stdint.h>

typedef __bf16 bf16;
typedef __attribute__((ext_vector_type(16))) __bf16 v16bf;
typedef __attribute__((ext_vector_type(8)))  float v8f;
typedef __attribute__((ext_vector_type(4)))  unsigned int v4u;
typedef __attribute__((ext_vector_type(4)))  unsigned short v4us;

#define EE 1024
#define SS 2048
#define BB 2
#define HH 16
#define HDD 64

static __device__ __forceinline__ int lane_id() { return threadIdx.x & 31; }

static __device__ __forceinline__ v8f wmma_bf16(v16bf a, v16bf b, v8f c) {
  // D = A(16x32 bf16) * B(32x16 bf16) + C(16x16 f32)
  return __builtin_amdgcn_wmma_f32_16x16x32_bf16(false, a, false, b, (short)0, c, false, false);
}

static __device__ __forceinline__ v8f zero_v8f() {
  v8f z = {0.f, 0.f, 0.f, 0.f, 0.f, 0.f, 0.f, 0.f};
  return z;
}

// Async global->LDS copy, 16B per lane, tracked by ASYNCcnt.
static __device__ __forceinline__ void async_b128(void* lds, const void* g) {
  asm volatile("global_load_async_to_lds_b128 %0, %1, off"
               :: "v"((uint32_t)(uintptr_t)lds), "v"(g) : "memory");
}
static __device__ __forceinline__ void wait_async0() {
  asm volatile("s_wait_asynccnt 0" ::: "memory");
}

// 16-lane butterfly reductions on the VALU via DPP16 (no LDS traffic).
// xor1 = quad_perm[1,0,3,2](0xB1), xor2 = quad_perm[2,3,0,1](0x4E),
// xor7 = row_half_mirror(0x141), xor15 = row_mirror(0x140).
template <int CTRL>
static __device__ __forceinline__ float dpp_mov(float x) {
  return __builtin_bit_cast(float,
      __builtin_amdgcn_update_dpp(0, __builtin_bit_cast(int, x), CTRL, 0xf, 0xf, true));
}
static __device__ __forceinline__ float red_max16(float v) {
  v = fmaxf(v, dpp_mov<0xB1>(v));
  v = fmaxf(v, dpp_mov<0x4E>(v));
  v = fmaxf(v, dpp_mov<0x141>(v));
  v = fmaxf(v, dpp_mov<0x140>(v));
  return v;
}
static __device__ __forceinline__ float red_sum16(float v) {
  v += dpp_mov<0xB1>(v);
  v += dpp_mov<0x4E>(v);
  v += dpp_mov<0x141>(v);
  v += dpp_mov<0x140>(v);
  return v;
}

// A-matrix fragment (16x32 bf16). sm -> row-major [16][ld] tile (ld multiple of 8).
static __device__ __forceinline__ v16bf load_frag_a(const bf16* sm, int ld) {
  const int l = lane_id();
  const bf16* p = sm + (l & 15) * ld + ((l >> 4) << 3);
  v16bf a;
  reinterpret_cast<v4u*>(&a)[0] = *reinterpret_cast<const v4u*>(p);       // K = kb..kb+7
  reinterpret_cast<v4u*>(&a)[1] = *reinterpret_cast<const v4u*>(p + 16);  // K = kb+16..kb+23
  return a;
}

// B-matrix fragment (32x16 bf16). sm -> N-major [16][ld] tile (row = n, contiguous k).
static __device__ __forceinline__ v16bf load_frag_b(const bf16* sm, int ld) {
  const int l = lane_id();
  const bf16* p = sm + (l & 15) * ld + ((l >> 4) << 3);
  v16bf b;
  reinterpret_cast<v4u*>(&b)[0] = *reinterpret_cast<const v4u*>(p);
  reinterpret_cast<v4u*>(&b)[1] = *reinterpret_cast<const v4u*>(p + 16);
  return b;
}

// ---------------------------------------------------------------------------
// fp32 -> bf16 (n % 1024 == 0)
// ---------------------------------------------------------------------------
__global__ void cvt_f32_bf16(const float* __restrict__ s, bf16* __restrict__ d, int n) {
  int i = (blockIdx.x * 256 + threadIdx.x) * 4;
  if (i < n) {
    float4 v = *reinterpret_cast<const float4*>(s + i);
    alignas(8) bf16 t[4] = {(bf16)v.x, (bf16)v.y, (bf16)v.z, (bf16)v.w};
    *reinterpret_cast<uint2*>(d + i) = *reinterpret_cast<const uint2*>(t);
  }
}

// ---------------------------------------------------------------------------
// fp32 [K][N] -> bf16 [N][K] (weight transpose, 32x32 tiles, 256 threads)
// ---------------------------------------------------------------------------
__global__ __launch_bounds__(256) void cvt_transpose(const float* __restrict__ w,
                                                     bf16* __restrict__ wt, int K, int N) {
  __shared__ float t[32][33];
  const int k0 = blockIdx.y * 32, n0 = blockIdx.x * 32;
  const int tx = threadIdx.x & 31, ty = threadIdx.x >> 5;
#pragma unroll
  for (int i = 0; i < 32; i += 8)
    t[ty + i][tx] = w[(size_t)(k0 + ty + i) * N + n0 + tx];
  __syncthreads();
#pragma unroll
  for (int i = 0; i < 32; i += 8)
    wt[(size_t)(n0 + ty + i) * K + k0 + tx] = (bf16)t[tx][ty + i];
}

// ---------------------------------------------------------------------------
// GEMM: C[M,N](bf16) = gelu?(A[M,K](bf16) * BT[N,K](bf16)^T + bias[N])
// block tile 64x256, BK=32, 256 threads = 8 waves (2m x 4n), wave tile 32x64.
// A/B tiles staged with global_load_async_to_lds_b128 (double buffered).
// ---------------------------------------------------------------------------
template <int GELU>
__global__ __launch_bounds__(256) void gemm_bf16(
    const bf16* __restrict__ A, const bf16* __restrict__ BT,
    const float* __restrict__ bias, bf16* __restrict__ C,
    int M, int N, int K) {
  constexpr int LDA = 40;  // 32 + 8 pad; row stride 80B (16B aligned)
  constexpr int LDB = 40;
  __shared__ bf16 As[2][64 * LDA];
  __shared__ bf16 Bs[2][256 * LDB];

  const int tid = threadIdx.x;
  const int m0 = blockIdx.y * 64;
  const int n0 = blockIdx.x * 256;

  auto stageA = [&](int buf, int k0) {
    // 64 rows x 32 k = 256 x 16B chunks, one async per thread
    int row = tid >> 2, c = (tid & 3) << 3;
    async_b128(&As[buf][row * LDA + c], A + (size_t)(m0 + row) * K + k0 + c);
  };
  auto stageB = [&](int buf, int k0) {
    // 256 rows(n) x 32 k = 1024 x 16B chunks, 4 asyncs per thread
#pragma unroll
    for (int i = 0; i < 4; ++i) {
      int idx = tid + i * 256;
      int row = idx >> 2, c = (idx & 3) << 3;
      async_b128(&Bs[buf][row * LDB + c], BT + (size_t)(n0 + row) * K + k0 + c);
    }
  };

  const int wave = tid >> 5;
  const int wm = wave >> 2, wn = wave & 3;

  v8f acc[2][4];
#pragma unroll
  for (int i = 0; i < 2; ++i)
#pragma unroll
    for (int j = 0; j < 4; ++j) acc[i][j] = zero_v8f();

  const int nk = K >> 5;
  stageA(0, 0);
  stageB(0, 0);
  wait_async0();
  __syncthreads();
  for (int kt = 0; kt < nk; ++kt) {
    const int cur = kt & 1;
    if (kt + 1 < nk) {  // overlap next tile's async copies with this tile's WMMAs
      stageA(cur ^ 1, (kt + 1) << 5);
      stageB(cur ^ 1, (kt + 1) << 5);
    }
    v16bf a0 = load_frag_a(&As[cur][(wm * 32) * LDA], LDA);
    v16bf a1 = load_frag_a(&As[cur][(wm * 32 + 16) * LDA], LDA);
#pragma unroll
    for (int j = 0; j < 4; ++j) {
      v16bf b = load_frag_b(&Bs[cur][(wn * 64 + j * 16) * LDB], LDB);
      acc[0][j] = wmma_bf16(a0, b, acc[0][j]);
      acc[1][j] = wmma_bf16(a1, b, acc[1][j]);
    }
    wait_async0();
    __syncthreads();
  }

  const int l = lane_id();
  const int colL = l & 15;
  const int rb = (l >> 4) << 3;
#pragma unroll
  for (int i = 0; i < 2; ++i) {
#pragma unroll
    for (int j = 0; j < 4; ++j) {
      const int col = n0 + wn * 64 + j * 16 + colL;
      const float bv = bias[col];
#pragma unroll
      for (int r = 0; r < 8; ++r) {
        const int row = m0 + wm * 32 + i * 16 + rb + r;
        float v = acc[i][j][r] + bv;
        if (GELU) v = 0.5f * v * (1.0f + erff(v * 0.70710678118654752f));
        C[(size_t)row * N + col] = (bf16)v;
      }
    }
  }
}

// ---------------------------------------------------------------------------
// Flash attention: one block = (b, h, 128-row Q tile), 256 threads = 8 waves,
// each wave owns 16 Q rows. K/V tiles of 64 keys streamed through LDS.
// qkv is bf16 [B*S][3E]; Q/K staged by async copies, V transposed manually.
// ---------------------------------------------------------------------------
__global__ __launch_bounds__(256) void attn_fa(const bf16* __restrict__ qkv,
                                               bf16* __restrict__ out) {
  constexpr int LD = 72;  // 64 + 8 pad; row stride 144B (16B aligned)
  __shared__ bf16 Qs[128 * LD];    // [q][d]
  __shared__ bf16 Ks[64 * LD];     // [key][d]
  __shared__ bf16 Vt[64 * LD];     // [d][key]
  __shared__ bf16 Ps[8][16 * LD];  // per-wave P staging [q][key]

  const int tid = threadIdx.x;
  const int q0 = blockIdx.x * 128;
  const int h = blockIdx.y;
  const int b = blockIdx.z;
  const size_t rowbase = (size_t)b * SS;
  const int qoff = h * HDD;
  const int koff = EE + h * HDD;
  const int voff = 2 * EE + h * HDD;

  // stage Q (128 x 64 bf16): 1024 x 16B chunks, 4 asyncs per thread
#pragma unroll
  for (int i = 0; i < 4; ++i) {
    int idx = tid + i * 256;
    int row = idx >> 3, c = (idx & 7) << 3;
    async_b128(&Qs[row * LD + c], qkv + (rowbase + q0 + row) * (3 * EE) + qoff + c);
  }

  const int wave = tid >> 5;
  const int l = lane_id();
  const int colL = l & 15;
  const int rb = (l >> 4) << 3;

  float mrun[8], lrun[8];
  v8f o[4];
#pragma unroll
  for (int r = 0; r < 8; ++r) { mrun[r] = -1e30f; lrun[r] = 0.f; }
#pragma unroll
  for (int n = 0; n < 4; ++n) o[n] = zero_v8f();

  for (int kt = 0; kt < SS / 64; ++kt) {
    const int k0 = kt * 64;
    // K tile: 64 x 64 bf16 = 512 x 16B chunks, 2 asyncs per thread
#pragma unroll
    for (int i = 0; i < 2; ++i) {
      int idx = tid + i * 256;
      int row = idx >> 3, c = (idx & 7) << 3;
      async_b128(&Ks[row * LD + c], qkv + (rowbase + k0 + row) * (3 * EE) + koff + c);
    }
    // V tile transposed: [d][key]
#pragma unroll
    for (int i = 0; i < 4; ++i) {
      int idx = tid + i * 256;
      int key = idx >> 4, d4 = (idx & 15) << 2;
      v4us vv = *reinterpret_cast<const v4us*>(qkv + (rowbase + k0 + key) * (3 * EE) + voff + d4);
      const bf16* e = reinterpret_cast<const bf16*>(&vv);
      Vt[(d4 + 0) * LD + key] = e[0];
      Vt[(d4 + 1) * LD + key] = e[1];
      Vt[(d4 + 2) * LD + key] = e[2];
      Vt[(d4 + 3) * LD + key] = e[3];
    }
    wait_async0();
    __syncthreads();

    // S = Q . K^T  (16 q-rows x 64 keys per wave), K-dim = HD = 64 -> 2 k-steps
    v8f s[4];
#pragma unroll
    for (int n = 0; n < 4; ++n) s[n] = zero_v8f();
    v16bf aq0 = load_frag_a(&Qs[wave * 16 * LD], LD);
    v16bf aq1 = load_frag_a(&Qs[wave * 16 * LD + 32], LD);
#pragma unroll
    for (int n = 0; n < 4; ++n) {
      v16bf bk0 = load_frag_b(&Ks[n * 16 * LD], LD);
      v16bf bk1 = load_frag_b(&Ks[n * 16 * LD + 32], LD);
      s[n] = wmma_bf16(aq0, bk0, s[n]);
      s[n] = wmma_bf16(aq1, bk1, s[n]);
    }

    // online softmax; row reductions entirely in VALU via DPP16
#pragma unroll
    for (int r = 0; r < 8; ++r) {
      float v0 = fmaxf(fmaxf(s[0][r], s[1][r]), fmaxf(s[2][r], s[3][r])) * 0.125f;
      v0 = red_max16(v0);
      const float mnew = fmaxf(mrun[r], v0);
      const float corr = __expf(mrun[r] - mnew);
      float rs = 0.f;
#pragma unroll
      for (int n = 0; n < 4; ++n) {
        float p = __expf(s[n][r] * 0.125f - mnew);
        s[n][r] = p;
        rs += p;
      }
      rs = red_sum16(rs);
      lrun[r] = lrun[r] * corr + rs;
      mrun[r] = mnew;
#pragma unroll
      for (int n = 0; n < 4; ++n) o[n][r] *= corr;
    }

    // P (C-layout regs) -> per-wave LDS -> A-layout frags; O += P . V
#pragma unroll
    for (int n = 0; n < 4; ++n)
#pragma unroll
      for (int r = 0; r < 8; ++r)
        Ps[wave][(rb + r) * LD + n * 16 + colL] = (bf16)s[n][r];
    asm volatile("" ::: "memory");  // same-wave LDS RAW ordering at IR level

    v16bf ap0 = load_frag_a(&Ps[wave][0], LD);
    v16bf ap1 = load_frag_a(&Ps[wave][32], LD);
#pragma unroll
    for (int n = 0; n < 4; ++n) {
      v16bf bv0 = load_frag_b(&Vt[n * 16 * LD], LD);
      v16bf bv1 = load_frag_b(&Vt[n * 16 * LD + 32], LD);
      o[n] = wmma_bf16(ap0, bv0, o[n]);
      o[n] = wmma_bf16(ap1, bv1, o[n]);
    }
    __syncthreads();
  }

  // write O / l (bf16) to [B*S][E] with head offset
#pragma unroll
  for (int n = 0; n < 4; ++n) {
#pragma unroll
    for (int r = 0; r < 8; ++r) {
      const size_t row = rowbase + q0 + wave * 16 + rb + r;
      out[row * EE + h * HDD + n * 16 + colL] = (bf16)(o[n][r] / lrun[r]);
    }
  }
}

// ---------------------------------------------------------------------------
// outf (f32) [+ outb (bf16)] = layernorm(res + y) * gamma + beta, E = 1024
// ---------------------------------------------------------------------------
__global__ __launch_bounds__(256) void residual_ln(const float* __restrict__ res,
                                                   const bf16* __restrict__ y,
                                                   const float* __restrict__ gamma,
                                                   const float* __restrict__ beta,
                                                   float* __restrict__ outf,
                                                   bf16* __restrict__ outb) {
  const int row = blockIdx.x;
  const int tid = threadIdx.x;
  const float* r = res + (size_t)row * EE;
  const bf16* yy = y + (size_t)row * EE;
  float x[4], s = 0.f, s2 = 0.f;
#pragma unroll
  for (int i = 0; i < 4; ++i) {
    int c = tid + i * 256;
    float v = r[c] + (float)yy[c];
    x[i] = v;
    s += v;
    s2 += v * v;
  }
#pragma unroll
  for (int off = 16; off > 0; off >>= 1) {
    s += __shfl_xor(s, off, 32);
    s2 += __shfl_xor(s2, off, 32);
  }
  __shared__ float rs[8], rs2[8];
  const int wave = tid >> 5, lane = tid & 31;
  if (lane == 0) { rs[wave] = s; rs2[wave] = s2; }
  __syncthreads();
  float ts = 0.f, ts2 = 0.f;
#pragma unroll
  for (int w = 0; w < 8; ++w) { ts += rs[w]; ts2 += rs2[w]; }
  const float mu = ts * (1.f / EE);
  const float var = ts2 * (1.f / EE) - mu * mu;
  const float rstd = rsqrtf(var + 1e-5f);
#pragma unroll
  for (int i = 0; i < 4; ++i) {
    int c = tid + i * 256;
    float o = (x[i] - mu) * rstd * gamma[c] + beta[c];
    outf[(size_t)row * EE + c] = o;
    if (outb) outb[(size_t)row * EE + c] = (bf16)o;
  }
}

// ---------------------------------------------------------------------------
extern "C" void kernel_launch(void* const* d_in, const int* in_sizes, int n_in,
                              void* d_out, int out_size, void* d_ws, size_t ws_size,
                              hipStream_t stream) {
  const float* x     = (const float*)d_in[0];
  const float* w_qkv = (const float*)d_in[1];
  const float* b_qkv = (const float*)d_in[2];
  const float* w_out = (const float*)d_in[3];
  const float* b_out = (const float*)d_in[4];
  const float* w_ff1 = (const float*)d_in[5];
  const float* b_ff1 = (const float*)d_in[6];
  const float* w_ff2 = (const float*)d_in[7];
  const float* b_ff2 = (const float*)d_in[8];
  const float* g1    = (const float*)d_in[9];
  const float* be1   = (const float*)d_in[10];
  const float* g2    = (const float*)d_in[11];
  const float* be2   = (const float*)d_in[12];
  float* out = (float*)d_out;

  char* ws = (char*)d_ws;
  const int M = BB * SS;  // 4096

  // pre-transposed bf16 weights [N][K]
  bf16* wqkvT = (bf16*)(ws);                       // 6291456 B
  bf16* woutT = (bf16*)(ws + 6291456);             // 2097152 B
  bf16* wff1T = (bf16*)(ws + 8388608);             // 8388608 B
  bf16* wff2T = (bf16*)(ws + 16777216);            // 8388608 B
  // bf16 activations
  bf16* x_bf    = (bf16*)(ws + 25165824);          // 4096x1024
  bf16* qkv_bf  = (bf16*)(ws + 33554432);          // 4096x3072
  bf16* attn_bf = (bf16*)(ws + 58720256);          // 4096x1024
  bf16* y1_bf   = (bf16*)(ws + 67108864);          // 4096x1024
  float* x1_f   = (float*)(ws + 75497472);         // 4096x1024 f32 (residual)
  bf16* x1_bf   = (bf16*)(ws + 92274688);          // 4096x1024
  bf16* h_bf    = qkv_bf;                          // reuse qkv+attn region: 4096x4096
  bf16* y2_bf   = y1_bf;                           // reuse

  cvt_f32_bf16<<<(M * EE) / 1024, 256, 0, stream>>>(x, x_bf, M * EE);
  cvt_transpose<<<dim3(3 * EE / 32, EE / 32), 256, 0, stream>>>(w_qkv, wqkvT, EE, 3 * EE);
  cvt_transpose<<<dim3(EE / 32, EE / 32), 256, 0, stream>>>(w_out, woutT, EE, EE);
  cvt_transpose<<<dim3(4 * EE / 32, EE / 32), 256, 0, stream>>>(w_ff1, wff1T, EE, 4 * EE);
  cvt_transpose<<<dim3(EE / 32, 4 * EE / 32), 256, 0, stream>>>(w_ff2, wff2T, 4 * EE, EE);

  // qkv = x @ w_qkv + b_qkv
  gemm_bf16<0><<<dim3(3 * EE / 256, M / 64), 256, 0, stream>>>(x_bf, wqkvT, b_qkv, qkv_bf, M, 3 * EE, EE);
  // attention
  attn_fa<<<dim3(SS / 128, HH, BB), 256, 0, stream>>>(qkv_bf, attn_bf);
  // y1 = attn @ w_out + b_out
  gemm_bf16<0><<<dim3(EE / 256, M / 64), 256, 0, stream>>>(attn_bf, woutT, b_out, y1_bf, M, EE, EE);
  // x1 = LN(x + y1)  (f32 residual copy + bf16 GEMM copy)
  residual_ln<<<M, 256, 0, stream>>>(x, y1_bf, g1, be1, x1_f, x1_bf);
  // h = gelu(x1 @ w_ff1 + b_ff1)
  gemm_bf16<1><<<dim3(4 * EE / 256, M / 64), 256, 0, stream>>>(x1_bf, wff1T, b_ff1, h_bf, M, 4 * EE, EE);
  // y2 = h @ w_ff2 + b_ff2
  gemm_bf16<0><<<dim3(EE / 256, M / 64), 256, 0, stream>>>(h_bf, wff2T, b_ff2, y2_bf, M, EE, 4 * EE);
  // out = LN(x1 + y2) (f32 only)
  residual_ln<<<M, 256, 0, stream>>>(x1_f, y2_bf, g2, be2, out, nullptr);
}
